// OuterProductMean_72249939853677
// MI455X (gfx1250) — compile-verified, matrix operand order checked by reading, not run on hardware
//
#include <hip/hip_runtime.h>
#include <cstdint>

typedef __attribute__((ext_vector_type(16))) _Float16 v16h;
typedef __attribute__((ext_vector_type(8)))  float    v8f;

#define S_LEN 384
#define N_LEN 512
#define C_DIM 32
#define CZ    128
#define K1    512              // GEMM1 reduction = N_LEN
#define R1    (S_LEN * C_DIM)  // 12288 rows of aT/bT

// ---------------------------------------------------------------------------
// Kernel 1: LayerNorm over C=32 (one wave32 per (s,n) row; lane == channel)
// + 64x32 projection, writing a^T and b^T in [i*32+c, n] f16 layout so both
// GEMM1 operands are K-contiguous.
// ---------------------------------------------------------------------------
__global__ __launch_bounds__(256) void opm_ln_proj(
    const float* __restrict__ m_si, const float* __restrict__ ln_g,
    const float* __restrict__ ln_b, const float* __restrict__ w_ab,
    _Float16* __restrict__ aT, _Float16* __restrict__ bT)
{
  __shared__ float wsm[64 * 32];   // w_ab, cached once per block
  __shared__ float msm[8][32];     // normalized row per wave
  const int tid  = threadIdx.x;
  const int lane = tid & 31;
  const int wv   = tid >> 5;

  for (int u = tid; u < 64 * 32; u += 256) wsm[u] = w_ab[u];

  const long row = (long)blockIdx.x * 8 + wv;   // 0 .. S*N-1
  const float x = m_si[row * 32 + lane];

  float s = x;
  #pragma unroll
  for (int o = 16; o > 0; o >>= 1) s += __shfl_xor(s, o, 32);
  const float mu = s * (1.0f / 32.0f);
  const float dx = x - mu;
  float v = dx * dx;
  #pragma unroll
  for (int o = 16; o > 0; o >>= 1) v += __shfl_xor(v, o, 32);
  const float rstd = rsqrtf(v * (1.0f / 32.0f) + 1e-5f);
  const float m = dx * rstd * ln_g[lane] + ln_b[lane];

  msm[wv][lane] = m;
  __syncthreads();   // wsm + msm visible

  float accA = 0.f, accB = 0.f;
  const float* wa = &wsm[lane * 32];          // d = lane       -> 'a'
  const float* wb = &wsm[(lane + 32) * 32];   // d = lane + 32  -> 'b'
  #pragma unroll
  for (int k = 0; k < 32; ++k) {
    const float mk = msm[wv][k];
    accA += mk * wa[k];
    accB += mk * wb[k];
  }

  const long i = row >> 9;        // seq index
  const long n = row & 511;       // reduction index
  aT[(i * 32 + lane) * (long)K1 + n] = (_Float16)accA;
  bT[(i * 32 + lane) * (long)K1 + n] = (_Float16)accB;
}

// ---------------------------------------------------------------------------
// Kernel 2: w_final f32 -> f16 (layout unchanged: [z, k], k-contiguous).
// ---------------------------------------------------------------------------
__global__ __launch_bounds__(256) void opm_cvt_wf(
    const float* __restrict__ w_final, _Float16* __restrict__ wfb)
{
  const int i = blockIdx.x * 256 + threadIdx.x;   // CZ*1024 = 131072 total
  wfb[i] = (_Float16)w_final[i];
}

// ---------------------------------------------------------------------------
// Kernel 3: fused  O = (1/N) * aT x bT^T  (128x128 tile, K=512)  then
//                  z = reshape(O) x w_final^T + bias  (16 pairs x 128, K=1024)
// 256 threads = 8 waves arranged 4x2 over the 8x8 grid of 16x16 O-tiles:
// each wave owns a 2x4 tile footprint (2 A-frags + 4 B-frags -> 8 WMMAs per
// k-step, 2x register reuse per fragment). O never touches HBM.
//
// Software pipeline: next k-step's global loads are issued before the 8-WMMA
// burst (pinned with a sched_barrier so the allocator keeps 4 distinct staging
// quads live across the burst) and retired into the alternate LDS buffer
// afterwards (one barrier per step). The 2x(A,B) stage buffers (32 KB) are
// time-aliased with the O rearrange buffer (32 KB) in a single smem block.
// ---------------------------------------------------------------------------
__global__ __launch_bounds__(256) void opm_fused_gemm(
    const _Float16* __restrict__ aT, const _Float16* __restrict__ bT,
    const _Float16* __restrict__ wfb, const float* __restrict__ b_final,
    float* __restrict__ out)
{
  __shared__ __align__(32) _Float16 smem[16384];  // 32 KB, dual purpose
  // Phase 1 layout: A buffers at 0 / 4096, B buffers at 8192 / 12288 (halves)
  // Phase 2 layout: O2[16][1024] over the whole block
  _Float16* const O2 = smem;

  const int tid  = threadIdx.x;
  const int lane = tid & 31;
  const int wave = tid >> 5;
  const int bi = blockIdx.x;          // i-block (4 seq rows -> 128 O-rows)
  const int bj = blockIdx.y;          // j-block
  const int r0 = bi * 128;
  const int c0 = bj * 128;

  const int fr = lane & 15;           // fragment row within 16
  const int fh = lane >> 4;           // fragment half (K-split)

  const int twr = (wave >> 1) * 2;    // wave's tile-row base (2 rows)
  const int twc = (wave & 1) * 4;     // wave's tile-col base (4 cols)

  // per-thread tile-load coordinates: two uint4 per tile per k-step
  const int row0 = tid >> 2;          // 0..63
  const int row1 = row0 + 64;
  const int seg  = tid & 3;
  const _Float16* const gA0 = aT + (size_t)(r0 + row0) * K1 + seg * 8;
  const _Float16* const gA1 = aT + (size_t)(r0 + row1) * K1 + seg * 8;
  const _Float16* const gB0 = bT + (size_t)(c0 + row0) * K1 + seg * 8;
  const _Float16* const gB1 = bT + (size_t)(c0 + row1) * K1 + seg * 8;
  const int l0 = row0 * 32 + seg * 8; // LDS offsets (halves)
  const int l1 = row1 * 32 + seg * 8;

  const v8f vzero = {0.f, 0.f, 0.f, 0.f, 0.f, 0.f, 0.f, 0.f};
  v8f acc[2][4];
  #pragma unroll
  for (int ta = 0; ta < 2; ++ta)
    #pragma unroll
    for (int tb = 0; tb < 4; ++tb) acc[ta][tb] = vzero;

  // -------- stage k-step 0 into buffer 0 --------
  {
    const uint4 a0 = *(const uint4*)gA0;
    const uint4 a1 = *(const uint4*)gA1;
    const uint4 b0 = *(const uint4*)gB0;
    const uint4 b1 = *(const uint4*)gB1;
    *(uint4*)&smem[l0]         = a0;
    *(uint4*)&smem[l1]         = a1;
    *(uint4*)&smem[8192 + l0]  = b0;
    *(uint4*)&smem[8192 + l1]  = b1;
  }
  __syncthreads();

  // -------- GEMM1 main loop: 16 K-steps of 32, double buffered --------
  for (int ks = 0; ks < 16; ++ks) {
    const int cur = ks & 1;
    _Float16* const Acur = smem + cur * 4096;
    _Float16* const Bcur = smem + 8192 + cur * 4096;

    uint4 a0 = {}, a1 = {}, b0 = {}, b1 = {};
    if (ks + 1 < 16) {                       // issue next-step loads early
      const size_t off = (size_t)(ks + 1) * 32;
      a0 = *(const uint4*)(gA0 + off);
      a1 = *(const uint4*)(gA1 + off);
      b0 = *(const uint4*)(gB0 + off);
      b1 = *(const uint4*)(gB1 + off);
    }
    if (ks + 2 < 16) {                       // gfx1250 global_prefetch_b8
      __builtin_prefetch(gA0 + (size_t)(ks + 2) * 32, 0, 1);
      __builtin_prefetch(gB0 + (size_t)(ks + 2) * 32, 0, 1);
    }
#if __has_builtin(__builtin_amdgcn_sched_barrier)
    // Pin the staged loads above the WMMA burst: forces 4 live staging quads
    // so their s_wait_loadcnt lands after the matrix work, not back-to-back.
    __builtin_amdgcn_sched_barrier(0);
#endif

    // 2 A-fragments + 4 B-fragments -> 8 WMMAs (each fragment used twice)
    const v16h af0 = *(const v16h*)&Acur[((twr + 0) * 16 + fr) * 32 + fh * 16];
    const v16h af1 = *(const v16h*)&Acur[((twr + 1) * 16 + fr) * 32 + fh * 16];
    #pragma unroll
    for (int tb = 0; tb < 4; ++tb) {
      const v16h bf = *(const v16h*)&Bcur[((twc + tb) * 16 + fr) * 32 + fh * 16];
      acc[0][tb] = __builtin_amdgcn_wmma_f32_16x16x32_f16(
          false, af0, false, bf, (short)0, acc[0][tb], false, false);
      acc[1][tb] = __builtin_amdgcn_wmma_f32_16x16x32_f16(
          false, af1, false, bf, (short)0, acc[1][tb], false, false);
    }

    if (ks + 1 < 16) {                       // retire loads into other buffer
      const int nxt = cur ^ 1;
      _Float16* const Anxt = smem + nxt * 4096;
      _Float16* const Bnxt = smem + 8192 + nxt * 4096;
      *(uint4*)&Anxt[l0] = a0;
      *(uint4*)&Anxt[l1] = a1;
      *(uint4*)&Bnxt[l0] = b0;
      *(uint4*)&Bnxt[l1] = b1;
    }
    __syncthreads();
  }

  // -------- rearrange O tile -> GEMM2 A operand (aliases stage buffers) ----
  // C/D layout: vgpr v, lanes 0-15 -> M=v, lanes 16-31 -> M=v+8; N = lane%16.
  const float inv_n = 1.0f / (float)N_LEN;
  #pragma unroll
  for (int ta = 0; ta < 2; ++ta) {
    #pragma unroll
    for (int tb = 0; tb < 4; ++tb) {
      #pragma unroll
      for (int v = 0; v < 8; ++v) {
        const int orow = (twr + ta) * 16 + v + ((lane & 16) ? 8 : 0); // i*32+c
        const int ocol = (twc + tb) * 16 + fr;                        // j*32+d
        const int p  = ((orow >> 5) << 2) | (ocol >> 5);              // pair
        const int k2 = ((orow & 31) << 5) | (ocol & 31);              // c*32+d
        O2[p * 1024 + k2] = (_Float16)(acc[ta][tb][v] * inv_n);
      }
    }
  }
  __syncthreads();

  // -------- GEMM2: M=16 pairs, N=128 (wave -> 16 cols), K=1024 --------
  v8f zacc = vzero;
  for (int ks = 0; ks < 32; ++ks) {
    const v16h a2 = *(const v16h*)&O2[fr * 1024 + ks * 32 + fh * 16];
    const v16h b2 = *(const v16h*)(wfb + (size_t)(wave * 16 + fr) * 1024
                                       + ks * 32 + fh * 16);
    zacc = __builtin_amdgcn_wmma_f32_16x16x32_f16(
        false, a2, false, b2, (short)0, zacc, false, false);
  }

  // -------- epilogue: bias + store z --------
  const int zc = wave * 16 + fr;
  const float bias = b_final[zc];
  #pragma unroll
  for (int v = 0; v < 8; ++v) {
    const int p = v + ((lane & 16) ? 8 : 0);   // pair index
    const int i = bi * 4 + (p >> 2);
    const int j = bj * 4 + (p & 3);
    out[((size_t)i * S_LEN + j) * CZ + zc] = zacc[v] + bias;
  }
}

// ---------------------------------------------------------------------------
extern "C" void kernel_launch(void* const* d_in, const int* in_sizes, int n_in,
                              void* d_out, int out_size, void* d_ws, size_t ws_size,
                              hipStream_t stream) {
  (void)in_sizes; (void)n_in; (void)out_size; (void)ws_size;
  const float* m_si    = (const float*)d_in[0];
  const float* ln_g    = (const float*)d_in[1];
  const float* ln_b    = (const float*)d_in[2];
  const float* w_ab    = (const float*)d_in[3];
  const float* w_final = (const float*)d_in[4];
  const float* b_final = (const float*)d_in[5];
  float* out = (float*)d_out;

  // workspace: aT (12288x512 f16) | bT (12288x512 f16) | wfb (128x1024 f16)
  _Float16* aT  = (_Float16*)d_ws;
  _Float16* bT  = aT + (size_t)R1 * K1;
  _Float16* wfb = bT + (size_t)R1 * K1;

  opm_ln_proj<<<(S_LEN * N_LEN) / 8, 256, 0, stream>>>(m_si, ln_g, ln_b, w_ab, aT, bT);
  opm_cvt_wf<<<(CZ * 1024) / 256, 256, 0, stream>>>(w_final, wfb);
  opm_fused_gemm<<<dim3(S_LEN / 4, S_LEN / 4), 256, 0, stream>>>(aT, bT, wfb, b_final, out);
}
